// CrossAttention_27874337751456
// MI455X (gfx1250) — matrix operand order repaired and checked by reading, not compiled, not run on hardware
//
#include <hip/hip_runtime.h>
#include <stdint.h>

#define NH    16
#define DH    64
#define SEQ   2048
#define DM    1024
#define NB    2

typedef __attribute__((ext_vector_type(16))) _Float16 v16h;
typedef __attribute__((ext_vector_type(8)))  _Float16 v8h;
typedef __attribute__((ext_vector_type(8)))  float    v8f;

// exact parameter type clang reports for the async-LDS builtin
typedef int v4i_ __attribute__((__vector_size__(4 * sizeof(int))));
typedef __attribute__((address_space(1))) v4i_* as1_v4i;
typedef __attribute__((address_space(3))) v4i_* as3_v4i;

#if defined(__has_builtin)
#if __has_builtin(__builtin_amdgcn_global_load_async_to_lds_b128)
#define HAVE_ASYNC_LDS 1
#endif
#endif
#ifndef HAVE_ASYNC_LDS
#define HAVE_ASYNC_LDS 0
#endif

// ---------------------------------------------------------------------------
// WMMA wrapper: D = A(16x32 f16) * B(32x16 f16) + C(16x16 f32)
// ---------------------------------------------------------------------------
__device__ __forceinline__ v8f wmma16(v16h a, v16h b, v8f c) {
  return __builtin_amdgcn_wmma_f32_16x16x32_f16(
      /*neg_a=*/false, a, /*neg_b=*/false, b,
      /*c_mod=*/(short)0, c, /*reuse_a=*/false, /*reuse_b=*/false);
}

// ---------------------------------------------------------------------------
// A operand: 16(M) x 32(K) row-major tile at p, leading dim ld (elements).
// ISA 7.12.2: lane l -> M=l&15 ; element e -> K = 16*(e>>3) + 8*(l>>4) + (e&7)
// ---------------------------------------------------------------------------
__device__ __forceinline__ v16h ldA_h(const _Float16* p, int ld) {
  const int l = threadIdx.x & 31;
  const _Float16* q = p + (l & 15) * ld + (l >> 4) * 8;
  v8h lo = *(const v8h*)(q);
  v8h hi = *(const v8h*)(q + 16);
  v16h a;
#pragma unroll
  for (int i = 0; i < 8; ++i) { a[i] = lo[i]; a[i + 8] = hi[i]; }
  return a;
}

// ---------------------------------------------------------------------------
// B operand: 32(K) x 16(N) tile, given row-major [N][K] at p (i.e. B^T),
// leading dim ld. lane l -> N=l&15 ; element e -> K = 16*(e>>3) + 8*(l>>4) + (e&7)
// ---------------------------------------------------------------------------
__device__ __forceinline__ v16h ldB_h(const _Float16* p, int ld) {
  const int l = threadIdx.x & 31;
  const _Float16* q = p + (l & 15) * ld + (l >> 4) * 8;
  v8h lo = *(const v8h*)(q);
  v8h hi = *(const v8h*)(q + 16);
  v16h b;
#pragma unroll
  for (int i = 0; i < 8; ++i) { b[i] = lo[i]; b[i + 8] = hi[i]; }
  return b;
}

// ---------------------------------------------------------------------------
// Async global -> LDS 16B copy (ASYNCcnt path on gfx1250), with a portable
// synchronous fallback. wait_async(n): wait until ASYNCcnt <= n.
// ---------------------------------------------------------------------------
__device__ __forceinline__ void async_cp16(const _Float16* g, _Float16* l) {
#if HAVE_ASYNC_LDS
  __builtin_amdgcn_global_load_async_to_lds_b128(
      (as1_v4i)(uintptr_t)g, (as3_v4i)(uintptr_t)l, 0, 0);
#else
  *(v8h*)l = *(const v8h*)g;
#endif
}

__device__ __forceinline__ void wait_async(int n) {
#if HAVE_ASYNC_LDS
#if __has_builtin(__builtin_amdgcn_s_wait_asynccnt)
  if (n == 0) __builtin_amdgcn_s_wait_asynccnt(0);
  else        __builtin_amdgcn_s_wait_asynccnt(1);
#else
  if (n == 0) asm volatile("s_wait_asynccnt 0x0" ::: "memory");
  else        asm volatile("s_wait_asynccnt 0x1" ::: "memory");
#endif
#endif
  (void)n;
}

// Stage one 64(N) x 32(K) f16 weight panel into LDS: 256 threads x 16B.
__device__ __forceinline__ void stage_panel(const _Float16* Bp, int k0,
                                            _Float16* dst) {
  const int t = threadIdx.x;        // 0..255
  const int row = t >> 2, c = t & 3;
  async_cp16(Bp + (size_t)row * DM + k0 + c * 8, dst + row * 32 + c * 8);
}

// ---------------------------------------------------------------------------
// Kernel 0: one-time f32 -> f16 conversion (x and the four weight matrices).
// ---------------------------------------------------------------------------
__global__ void __launch_bounds__(256) cvt_f16(
    const float* __restrict__ src, _Float16* __restrict__ dst, int n) {
  const int i = (blockIdx.x * 256 + threadIdx.x) * 8;
  if (i >= n) return;
  const float4 f0 = *(const float4*)(src + i);
  const float4 f1 = *(const float4*)(src + i + 4);
  v8h h;
  h[0] = (_Float16)f0.x; h[1] = (_Float16)f0.y;
  h[2] = (_Float16)f0.z; h[3] = (_Float16)f0.w;
  h[4] = (_Float16)f1.x; h[5] = (_Float16)f1.y;
  h[6] = (_Float16)f1.z; h[7] = (_Float16)f1.w;
  *(v8h*)(dst + i) = h;
}

// ---------------------------------------------------------------------------
// Kernel 1: QKV projections, all-f16 operands.  y = x @ W^T
// grid = (SEQ/256, NB*NH, 3), block = 256 (8 waves).
// Wave: M=32 x N=64. Weight panel shared by all waves -> double-buffered
// LDS staging via async global->LDS copies.
// acc[nt]   : rows m0..m0+15,  acc[4+nt] : rows m0+16..m0+31
// ---------------------------------------------------------------------------
__global__ void __launch_bounds__(256) qkv_proj(
    const _Float16* __restrict__ xh, const _Float16* __restrict__ Wqh,
    const _Float16* __restrict__ Wkh, const _Float16* __restrict__ Wvh,
    _Float16* __restrict__ Qh, _Float16* __restrict__ Kh,
    _Float16* __restrict__ Vt) {
  __shared__ __align__(16) _Float16 ldsB[2 * 64 * 32];  // 8 KB double buffer

  const int mat  = blockIdx.z;                 // 0=Q 1=K 2=V
  const int bh   = blockIdx.y;
  const int b    = bh >> 4, h = bh & 15;
  const int wave = threadIdx.x >> 5;
  const int m0   = blockIdx.x * 256 + wave * 32;

  const _Float16* W  = (mat == 0) ? Wqh : (mat == 1) ? Wkh : Wvh;
  const _Float16* Ap = xh + (size_t)(b * SEQ + m0) * DM;
  const _Float16* Bp = W + (size_t)(h * DH) * DM;

  v8f acc[8] = {};

  stage_panel(Bp, 0, ldsB);
  for (int i = 0; i < DM / 32; ++i) {
    const int k0 = i * 32;
    const bool more = (i + 1) < DM / 32;
    if (more) stage_panel(Bp, k0 + 32, ldsB + ((i + 1) & 1) * 2048);
    wait_async(more ? 1 : 0);
    __syncthreads();
    const _Float16* Bl = ldsB + (i & 1) * 2048;
    const v16h a0 = ldA_h(Ap + k0, DM);
    const v16h a1 = ldA_h(Ap + (size_t)16 * DM + k0, DM);
#pragma unroll
    for (int nt = 0; nt < 4; ++nt) {
      const v16h bt = ldB_h(Bl + nt * 16 * 32, 32);
      acc[nt]     = wmma16(a0, bt, acc[nt]);
      acc[4 + nt] = wmma16(a1, bt, acc[4 + nt]);
    }
    __syncthreads();
  }

  const int l = threadIdx.x & 31, n = l & 15, kb = l >> 4;
  const float scale = (mat == 0) ? 0.18033688f : 1.0f; // 1/sqrt(64) * log2(e)
#pragma unroll
  for (int t = 0; t < 8; ++t) {
    const int nt = t & 3;
    const int half = t >> 2;
#pragma unroll
    for (int j = 0; j < 8; ++j) {
      const float v = acc[t][j] * scale;
      const int row = m0 + half * 16 + j + 8 * kb;  // token index
      const int col = nt * 16 + n;                  // dh index
      if (mat == 0)       Qh[(size_t)(bh * SEQ + row) * DH + col] = (_Float16)v;
      else if (mat == 1)  Kh[(size_t)(bh * SEQ + row) * DH + col] = (_Float16)v;
      else                Vt[(size_t)(bh * DH + col) * SEQ + row] = (_Float16)v;
    }
  }
}

// ---------------------------------------------------------------------------
// Kernel 2: flash attention.  grid = (SEQ/64, NB*NH), block = 128 (4 waves).
// Per-wave private LDS P-tile; per-wave DS ordering makes barriers
// unnecessary (no cross-wave LDS sharing).
// ---------------------------------------------------------------------------
__global__ void __launch_bounds__(128) attn_fwd(
    const _Float16* __restrict__ Qh, const _Float16* __restrict__ Kh,
    const _Float16* __restrict__ Vt, _Float16* __restrict__ Oh) {
  __shared__ __align__(16) _Float16 ldsP[4 * 16 * 64];

  const int bh   = blockIdx.y;
  const int b    = bh >> 4, h = bh & 15;
  const int wave = threadIdx.x >> 5;
  const int q0   = blockIdx.x * 64 + wave * 16;
  const int l    = threadIdx.x & 31, n = l & 15, kb = l >> 4;
  _Float16* P = ldsP + wave * (16 * 64);

  const _Float16* Qp = Qh + (size_t)(bh * SEQ + q0) * DH;
  const v16h qa0 = ldA_h(Qp, DH);
  const v16h qa1 = ldA_h(Qp + 32, DH);

  v8f o[4] = {};
  float mrow[8], lrow[8];
#pragma unroll
  for (int j = 0; j < 8; ++j) { mrow[j] = -1e30f; lrow[j] = 0.0f; }

  for (int j0 = 0; j0 < SEQ; j0 += 64) {
    // S = Q @ K^T (scaled; log2e already folded into Q)
    v8f s[4] = {};
#pragma unroll
    for (int nt = 0; nt < 4; ++nt) {
      const _Float16* Kp = Kh + (size_t)(bh * SEQ + j0 + nt * 16) * DH;
      s[nt] = wmma16(qa0, ldB_h(Kp, DH), s[nt]);
      s[nt] = wmma16(qa1, ldB_h(Kp + 32, DH), s[nt]);
    }
    // online softmax: each row lives across 16 lanes of a half-wave
#pragma unroll
    for (int j = 0; j < 8; ++j) {
      float rm = fmaxf(fmaxf(s[0][j], s[1][j]), fmaxf(s[2][j], s[3][j]));
      rm = fmaxf(rm, __shfl_xor(rm, 1, 32));
      rm = fmaxf(rm, __shfl_xor(rm, 2, 32));
      rm = fmaxf(rm, __shfl_xor(rm, 4, 32));
      rm = fmaxf(rm, __shfl_xor(rm, 8, 32));
      const float mnew = fmaxf(mrow[j], rm);
      const float corr = __builtin_amdgcn_exp2f(mrow[j] - mnew);
      mrow[j] = mnew;
      float ps = 0.0f;
#pragma unroll
      for (int nt = 0; nt < 4; ++nt) {
        const float p = __builtin_amdgcn_exp2f(s[nt][j] - mnew);
        ps += p;
        o[nt][j] *= corr;
        P[(j + 8 * kb) * 64 + nt * 16 + n] = (_Float16)p;
      }
      ps += __shfl_xor(ps, 1, 32);
      ps += __shfl_xor(ps, 2, 32);
      ps += __shfl_xor(ps, 4, 32);
      ps += __shfl_xor(ps, 8, 32);
      lrow[j] = lrow[j] * corr + ps;
    }
    // O += P @ V ; per-wave DS ordering guarantees the P stores above are
    // visible to these loads (same wave, in-order LDS pipeline).
    const v16h pa0 = ldA_h(P, 64);
    const v16h pa1 = ldA_h(P + 32, 64);
#pragma unroll
    for (int nt = 0; nt < 4; ++nt) {
      const _Float16* Vp = Vt + (size_t)(bh * DH + nt * 16) * SEQ + j0;
      o[nt] = wmma16(pa0, ldB_h(Vp, SEQ), o[nt]);
      o[nt] = wmma16(pa1, ldB_h(Vp + 32, SEQ), o[nt]);
    }
  }

  // normalize and store [b][n][h*dh] so the out-projection reads row-major
#pragma unroll
  for (int j = 0; j < 8; ++j) {
    const float inv = 1.0f / lrow[j];
    const int row = q0 + j + 8 * kb;
#pragma unroll
    for (int nt = 0; nt < 4; ++nt)
      Oh[(size_t)(b * SEQ + row) * DM + h * DH + nt * 16 + n] =
          (_Float16)(o[nt][j] * inv);
  }
}

// ---------------------------------------------------------------------------
// Kernel 3: output projection  out = Oh @ Wo^T + bo   (f32 out)
// grid = (NB*SEQ/256, DM/64), block = 256 (8 waves); M=32 x N=64 per wave.
// Same async LDS staging of the shared weight panel.
// ---------------------------------------------------------------------------
__global__ void __launch_bounds__(256) out_proj(
    const _Float16* __restrict__ Oh, const _Float16* __restrict__ Woh,
    const float* __restrict__ bo, float* __restrict__ out) {
  __shared__ __align__(16) _Float16 ldsB[2 * 64 * 32];  // 8 KB double buffer

  const int wave = threadIdx.x >> 5;
  const int m0 = blockIdx.x * 256 + wave * 32;
  const int n0 = blockIdx.y * 64;

  const _Float16* Ap = Oh + (size_t)m0 * DM;
  const _Float16* Bp = Woh + (size_t)n0 * DM;
  v8f acc[8] = {};

  stage_panel(Bp, 0, ldsB);
  for (int i = 0; i < DM / 32; ++i) {
    const int k0 = i * 32;
    const bool more = (i + 1) < DM / 32;
    if (more) stage_panel(Bp, k0 + 32, ldsB + ((i + 1) & 1) * 2048);
    wait_async(more ? 1 : 0);
    __syncthreads();
    const _Float16* Bl = ldsB + (i & 1) * 2048;
    const v16h a0 = ldA_h(Ap + k0, DM);
    const v16h a1 = ldA_h(Ap + (size_t)16 * DM + k0, DM);
#pragma unroll
    for (int nt = 0; nt < 4; ++nt) {
      const v16h bt = ldB_h(Bl + nt * 16 * 32, 32);
      acc[nt]     = wmma16(a0, bt, acc[nt]);
      acc[4 + nt] = wmma16(a1, bt, acc[4 + nt]);
    }
    __syncthreads();
  }

  const int l = threadIdx.x & 31, n = l & 15, kb = l >> 4;
#pragma unroll
  for (int nt = 0; nt < 4; ++nt) {
    const int col = n0 + nt * 16 + n;
    const float bias = bo[col];
#pragma unroll
    for (int j = 0; j < 8; ++j) {
      const int row0 = m0 + j + 8 * kb;
      out[(size_t)row0 * DM + col] = acc[nt][j] + bias;
      out[(size_t)(row0 + 16) * DM + col] = acc[4 + nt][j] + bias;
    }
  }
}

// ---------------------------------------------------------------------------
extern "C" void kernel_launch(void* const* d_in, const int* in_sizes, int n_in,
                              void* d_out, int out_size, void* d_ws, size_t ws_size,
                              hipStream_t stream) {
  (void)in_sizes; (void)n_in; (void)out_size; (void)ws_size;
  const float* x  = (const float*)d_in[0];
  const float* Wq = (const float*)d_in[1];
  const float* Wk = (const float*)d_in[2];
  const float* Wv = (const float*)d_in[3];
  const float* Wo = (const float*)d_in[4];
  const float* bo = (const float*)d_in[5];
  float* out = (float*)d_out;

  const size_t MB = 1024 * 1024;
  char* ws = (char*)d_ws;
  _Float16* xh  = (_Float16*)(ws);              //  8 MB: x as f16
  _Float16* Wqh = (_Float16*)(ws + 8  * MB);    //  2 MB
  _Float16* Wkh = (_Float16*)(ws + 10 * MB);    //  2 MB
  _Float16* Wvh = (_Float16*)(ws + 12 * MB);    //  2 MB
  _Float16* Woh = (_Float16*)(ws + 14 * MB);    //  2 MB
  _Float16* Qh  = (_Float16*)(ws + 16 * MB);    //  8 MB
  _Float16* Kh  = (_Float16*)(ws + 24 * MB);    //  8 MB
  _Float16* Vt  = (_Float16*)(ws + 32 * MB);    //  8 MB
  _Float16* Oh  = (_Float16*)(ws + 40 * MB);    //  8 MB

  const int nx = NB * SEQ * DM;  // 4.19M
  const int nw = DM * DM;        // 1.05M
  cvt_f16<<<nx / (8 * 256), 256, 0, stream>>>(x, xh, nx);
  cvt_f16<<<nw / (8 * 256), 256, 0, stream>>>(Wq, Wqh, nw);
  cvt_f16<<<nw / (8 * 256), 256, 0, stream>>>(Wk, Wkh, nw);
  cvt_f16<<<nw / (8 * 256), 256, 0, stream>>>(Wv, Wvh, nw);
  cvt_f16<<<nw / (8 * 256), 256, 0, stream>>>(Wo, Woh, nw);

  dim3 g1(SEQ / 256, NB * NH, 3);
  qkv_proj<<<g1, 256, 0, stream>>>(xh, Wqh, Wkh, Wvh, Qh, Kh, Vt);

  dim3 g2(SEQ / 64, NB * NH);
  attn_fwd<<<g2, 128, 0, stream>>>(Qh, Kh, Vt, Oh);

  dim3 g3((NB * SEQ) / 256, DM / 64);
  out_proj<<<g3, 256, 0, stream>>>(Oh, Woh, bo, out);
}
